// SPINN3D_4612794876656
// MI455X (gfx1250) — matrix operand order repaired and checked by reading, not compiled, not run on hardware
//
#include <hip/hip_runtime.h>
#include <math.h>

typedef __attribute__((ext_vector_type(16))) _Float16 v16h;
typedef __attribute__((ext_vector_type(8)))  _Float16 v8h;
typedef __attribute__((ext_vector_type(8)))  float    v8f;

#define F    128
#define NPTS 320

// ---------------------------------------------------------------------------
// Phase 1: tiny MLP  v(1) -> tanh(128) -> tanh(128) -> linear(128)
// One 128-thread block per point; thread t owns feature t.
// Writes point-major f16 table: outh[point][feature]  ([320][128]).
// ---------------------------------------------------------------------------
__global__ __launch_bounds__(128)
void spinn_mlp_kernel(const float* __restrict__ vin,
                      const float* __restrict__ W1, const float* __restrict__ b1,
                      const float* __restrict__ W2, const float* __restrict__ b2,
                      const float* __restrict__ W3, const float* __restrict__ b3,
                      _Float16* __restrict__ outh) {
    __shared__ float h1[F];
    __shared__ float h2[F];
    const int t = threadIdx.x;
    const int p = blockIdx.x;

    const float v = vin[p];
    h1[t] = tanhf(fmaf(v, W1[t], b1[t]));      // W1 is (1,128)
    __syncthreads();

    float acc = b2[t];
    #pragma unroll 8
    for (int j = 0; j < F; ++j)                // column read: coalesced across t
        acc = fmaf(h1[j], W2[j * F + t], acc);
    h2[t] = tanhf(acc);
    __syncthreads();

    acc = b3[t];
    #pragma unroll 8
    for (int j = 0; j < F; ++j)
        acc = fmaf(h2[j], W3[j * F + t], acc);

    outh[p * F + t] = (_Float16)acc;
}

// ---------------------------------------------------------------------------
// Phase 2: out[z,y,x] = sum_f fz[z,f]*fy[y,f]*fx[x,f]
// GEMM view: A[(z,y),f] = fz*fy (built in-register), B[f,x] = fx, K=128.
// Grid: (20 y-tiles, 320 z). Block: 128 threads = 4 waves; each wave owns
// 5 x-tiles of 16. v_wmma_f32_16x16x32_f16, f32 accumulate.
//
// A-fragment 16x32 f16 layout (ISA 7.12.2): lane in [0,15]: M=lane, halves
// e=0..7 -> K=e, e=8..15 -> K=e+8; lanes [16,31]: same M, K offset +8.
// B-fragment 32x16 f16: lane L: N=L%16, half e -> K = 8*(L/16) + (e<8 ? e : e+8).
// C/D 16x16 f32: VGPR v: M = v + 8*(L/16), N = L%16.
// ---------------------------------------------------------------------------
__global__ __launch_bounds__(128)
void spinn_contract_kernel(const _Float16* __restrict__ fzh,  // [320][128]
                           const _Float16* __restrict__ fyh,  // [320][128]
                           const _Float16* __restrict__ fxh,  // [320][128]
                           float* __restrict__ out) {         // [320][320][320]
    const int z    = blockIdx.y;
    const int y0   = blockIdx.x * 16;
    const int wave = threadIdx.x >> 5;
    const int lane = threadIdx.x & 31;
    const int r    = lane & 15;   // row-in-tile / column-in-tile index
    const int grp  = lane >> 4;   // half-wave group

    const _Float16* fzrow = fzh + z * F;
    const _Float16* fyrow = fyh + (y0 + r) * F;
    const int x_base = wave * 5 * 16;          // this wave's 5 x-tiles

    v8f acc[5] = {};

    #pragma unroll
    for (int kb = 0; kb < 4; ++kb) {           // K = 128 in steps of 32
        const int offA = kb * 32 + grp * 8;
        // A fragment: a[e] = fz[k]*fy[k,y] with k = offA + (e<8 ? e : e+8)
        const v8h az0 = *(const v8h*)(fzrow + offA);
        const v8h az1 = *(const v8h*)(fzrow + offA + 16);
        const v8h ay0 = *(const v8h*)(fyrow + offA);
        const v8h ay1 = *(const v8h*)(fyrow + offA + 16);
        const v8h a0 = az0 * ay0;              // v_pk_mul_f16
        const v8h a1 = az1 * ay1;
        const v16h a = __builtin_shufflevector(a0, a1,
                0, 1, 2, 3, 4, 5, 6, 7, 8, 9, 10, 11, 12, 13, 14, 15);

        const int offB = kb * 32 + grp * 8;
        #pragma unroll
        for (int i = 0; i < 5; ++i) {
            const int x = x_base + i * 16 + r;
            // two 8-half runs at offB and offB+16 (16B aligned): two global_load_b128
            const v8h b0 = *(const v8h*)(fxh + x * F + offB);
            const v8h b1 = *(const v8h*)(fxh + x * F + offB + 16);
            const v16h b = __builtin_shufflevector(b0, b1,
                    0, 1, 2, 3, 4, 5, 6, 7, 8, 9, 10, 11, 12, 13, 14, 15);
            acc[i] = __builtin_amdgcn_wmma_f32_16x16x32_f16(
                false, a, false, b, (short)0, acc[i], false, false);
        }
    }

    // Store: streaming 131MB output -> non-temporal to keep L2 clean.
    float* outz = out + (size_t)z * NPTS * NPTS;
    #pragma unroll
    for (int i = 0; i < 5; ++i) {
        const int x = x_base + i * 16 + r;
        #pragma unroll
        for (int v = 0; v < 8; ++v) {
            const int yy = y0 + v + grp * 8;
            __builtin_nontemporal_store(acc[i][v],
                                        &outz[(size_t)yy * NPTS + x]);
        }
    }
}

// ---------------------------------------------------------------------------
// d_in leaf order (pytree-flattened setup_inputs dict):
//  [0]=z [1]=y [2]=x
//  [3..8]   params_z: W1,b1,W2,b2,W3,b3
//  [9..14]  params_y
//  [15..20] params_x
// ---------------------------------------------------------------------------
extern "C" void kernel_launch(void* const* d_in, const int* in_sizes, int n_in,
                              void* d_out, int out_size, void* d_ws, size_t ws_size,
                              hipStream_t stream) {
    (void)in_sizes; (void)n_in; (void)out_size; (void)ws_size;

    const float* zin = (const float*)d_in[0];
    const float* yin = (const float*)d_in[1];
    const float* xin = (const float*)d_in[2];

    _Float16* fzh = (_Float16*)d_ws;               // [320][128] each, f16
    _Float16* fyh = fzh + NPTS * F;
    _Float16* fxh = fyh + NPTS * F;

    const float** pz = (const float**)&d_in[3];
    const float** py = (const float**)&d_in[9];
    const float** px = (const float**)&d_in[15];

    spinn_mlp_kernel<<<NPTS, 128, 0, stream>>>(
        zin, pz[0], pz[1], pz[2], pz[3], pz[4], pz[5], fzh);
    spinn_mlp_kernel<<<NPTS, 128, 0, stream>>>(
        yin, py[0], py[1], py[2], py[3], py[4], py[5], fyh);
    spinn_mlp_kernel<<<NPTS, 128, 0, stream>>>(
        xin, px[0], px[1], px[2], px[3], px[4], px[5], fxh);

    // (20 y-tiles, 320 z) workgroups; 4 waves each covering 320 x.
    spinn_contract_kernel<<<dim3(20, NPTS), 128, 0, stream>>>(
        fzh, fyh, fxh, (float*)d_out);
}
